// RNN_22883585753135
// MI455X (gfx1250) — matrix operand order repaired and checked by reading, not compiled, not run on hardware
//
#include <hip/hip_runtime.h>
#include <math.h>

typedef __attribute__((ext_vector_type(16))) _Float16 v16h;
typedef __attribute__((ext_vector_type(8)))  _Float16 v8h;
typedef __attribute__((ext_vector_type(8)))  float    v8f;
typedef __attribute__((ext_vector_type(4)))  float    v4f;

#define Hdim  128
#define Tdim  4096
#define Bdim  64
#define ROWS  16      // batch rows per workgroup
#define PITCH 136     // f16 elements per LDS row (272B => 4-bank rotation per row)

// ---- tanh: native V_TANH_F32 if this clang exposes it; else branch-free exp2/rcp.
// Host pass only needs to parse a valid body (never executed on host).
#if defined(__AMDGCN__)
  #if __has_builtin(__builtin_amdgcn_tanhf)
__device__ __forceinline__ float fast_tanh(float x) { return __builtin_amdgcn_tanhf(x); }
  #elif __has_builtin(__builtin_amdgcn_tanh_f32)
__device__ __forceinline__ float fast_tanh(float x) { return __builtin_amdgcn_tanh_f32(x); }
  #else
__device__ __forceinline__ float fast_tanh(float x) {
  float ax = __builtin_fabsf(x);
  float e  = __builtin_amdgcn_exp2f(ax * 2.8853900817779268f);  // 2^(2|x|*log2e)
  float r  = 1.0f - 2.0f * __builtin_amdgcn_rcpf(e + 1.0f);     // |x| big -> e=inf -> r=1
  return __builtin_copysignf(r, x);
}
  #endif
#else
__device__ __forceinline__ float fast_tanh(float x) { return tanhf(x); }
#endif

// 32x16 f16 B-fragment from row-major [N=128][K=128] f32 weights.
// Lane L: column n, k = kbase..kbase+15 (kbase = 32*kk + 16*(L>=16)).
__device__ __forceinline__ v16h bfrag_from_global(const float* __restrict__ W, int n, int kbase) {
  const v4f* p = (const v4f*)(W + n * Hdim + kbase);
  v4f q0 = p[0], q1 = p[1], q2 = p[2], q3 = p[3];
  v16h r;
#pragma unroll
  for (int j = 0; j < 4; ++j) {
    r[j]      = (_Float16)q0[j];
    r[4 + j]  = (_Float16)q1[j];
    r[8 + j]  = (_Float16)q2[j];
    r[12 + j] = (_Float16)q3[j];
  }
  return r;
}

// 16x32 f16 A-fragment from LDS tile [16][PITCH].
// Lane L: row m=L&15; j=0..15 -> k=32kk+kh+j, kh=16*(L>=16).
__device__ __forceinline__ v16h afrag_from_lds(const _Float16* S, int lane, int kk) {
  int m  = lane & 15;
  int kh = (lane >> 4) << 4;
  const _Float16* p = S + m * PITCH + kk * 32 + kh;
  v8h lo = *(const v8h*)p;
  v8h hi = *(const v8h*)(p + 8);
  return __builtin_shufflevector(lo, hi, 0,1,2,3,4,5,6,7,8,9,10,11,12,13,14,15);
}

__global__ __launch_bounds__(256, 1)
void rnn2_pipelined_scan(const float* __restrict__ x,
                         const float* __restrict__ Wih0, const float* __restrict__ Whh0,
                         const float* __restrict__ bih0, const float* __restrict__ bhh0,
                         const float* __restrict__ Wih1, const float* __restrict__ Whh1,
                         const float* __restrict__ bih1, const float* __restrict__ bhh1,
                         float* __restrict__ out)
{
  __shared__ __align__(16) _Float16 xs [2][ROWS * PITCH];
  __shared__ __align__(16) _Float16 h0s[2][ROWS * PITCH];
  __shared__ __align__(16) _Float16 h1s[2][ROWS * PITCH];

  const int tid   = threadIdx.x;
  const int lane  = tid & 31;
  const int wv    = tid >> 5;
  const int layer = wv >> 2;            // waves 0-3: layer 0 @ step s; waves 4-7: layer 1 @ step s-1
  const int g     = wv & 3;             // 32-column slab within the layer
  const int b0    = blockIdx.x * ROWS;  // this WG's batch rows

  const float* Wih = layer ? Wih1 : Wih0;
  const float* Whh = layer ? Whh1 : Whh0;
  const float* bih = layer ? bih1 : bih0;
  const float* bhh = layer ? bhh1 : bhh0;

  // ---- register-resident weight B-fragments: 2 column tiles x 4 k-steps x 2 matrices ----
  v16h fih[2][4], fhh[2][4];
  int n_[2];
  float bias[2];
  {
    const int kh16 = (lane >> 4) << 4;  // 0 or 16
#pragma unroll
    for (int tne = 0; tne < 2; ++tne) {
      const int n = g * 32 + tne * 16 + (lane & 15);
      n_[tne]   = n;
      bias[tne] = bih[n] + bhh[n];
#pragma unroll
      for (int kk = 0; kk < 4; ++kk) {
        fih[tne][kk] = bfrag_from_global(Wih, n, kk * 32 + kh16);
        fhh[tne][kk] = bfrag_from_global(Whh, n, kk * 32 + kh16);
      }
    }
  }
  const int rbase = (lane >> 4) << 3;   // C-tile row base for this lane half

  // ---- init: zero h state (h0[-1], h1[-1], h1[-2]), stage x_0 ----
  for (int i = tid; i < ROWS * PITCH; i += 256) {
    h0s[0][i] = (_Float16)0.f;
    h1s[0][i] = (_Float16)0.f;
    h1s[1][i] = (_Float16)0.f;
  }
  const int xm = tid >> 4;          // staging row 0..15
  const int xc = (tid & 15) * 8;    // staging col chunk
  {
    const float* xp = x + ((b0 + xm) * Tdim + 0) * Hdim + xc;
    v4f a0 = ((const v4f*)xp)[0], a1 = ((const v4f*)xp)[1];
    _Float16* d = &xs[0][xm * PITCH + xc];
#pragma unroll
    for (int j = 0; j < 4; ++j) { d[j] = (_Float16)a0[j]; d[4 + j] = (_Float16)a1[j]; }
  }
  __syncthreads();

  float* hid = out + Bdim * Tdim * Hdim;   // hidden_state [2][B][H]

  // Pipelined scan: iteration s produces h0[s] (group A) and h1[s-1] (group B).
  for (int s = 0; s <= Tdim; ++s) {
    const int p = s & 1;
    const int q = p ^ 1;

    // fetch x_{s+1} into registers (overlaps the WMMA work below)
    v4f xq0 = {}, xq1 = {};
    if (s + 1 < Tdim) {
      const float* xp = x + ((b0 + xm) * Tdim + (s + 1)) * Hdim + xc;
      xq0 = ((const v4f*)xp)[0];
      xq1 = ((const v4f*)xp)[1];
    }
    if (s + 8 < Tdim)
      __builtin_prefetch(x + ((b0 + xm) * Tdim + (s + 8)) * Hdim + xc, 0, 3);

    if (layer == 0) {
      if (s < Tdim) {
        // h0[s] = tanh(x[s] @ Wih0^T + h0[s-1] @ Whh0^T + b)
        v16h ax[4], ah[4];
#pragma unroll
        for (int kk = 0; kk < 4; ++kk) {
          ax[kk] = afrag_from_lds(xs[p],  lane, kk);
          ah[kk] = afrag_from_lds(h0s[p], lane, kk);
        }
        v8f c0 = {}, c1 = {};
#pragma unroll
        for (int kk = 0; kk < 4; ++kk) {
          c0 = __builtin_amdgcn_wmma_f32_16x16x32_f16(false, ax[kk], false, fih[0][kk], (short)0, c0, false, false);
          c1 = __builtin_amdgcn_wmma_f32_16x16x32_f16(false, ax[kk], false, fih[1][kk], (short)0, c1, false, false);
        }
#pragma unroll
        for (int kk = 0; kk < 4; ++kk) {
          c0 = __builtin_amdgcn_wmma_f32_16x16x32_f16(false, ah[kk], false, fhh[0][kk], (short)0, c0, false, false);
          c1 = __builtin_amdgcn_wmma_f32_16x16x32_f16(false, ah[kk], false, fhh[1][kk], (short)0, c1, false, false);
        }
#pragma unroll
        for (int r = 0; r < 8; ++r) {
          const int m = rbase + r;
          float v0 = fast_tanh(c0[r] + bias[0]);
          float v1 = fast_tanh(c1[r] + bias[1]);
          h0s[q][m * PITCH + n_[0]] = (_Float16)v0;
          h0s[q][m * PITCH + n_[1]] = (_Float16)v1;
          if (s == Tdim - 1) {                      // h_n layer 0
            hid[(b0 + m) * Hdim + n_[0]] = v0;
            hid[(b0 + m) * Hdim + n_[1]] = v1;
          }
        }
      }
    } else {
      if (s >= 1) {
        // h1[s-1] = tanh(h0[s-1] @ Wih1^T + h1[s-2] @ Whh1^T + b)
        v16h ax[4], ah[4];
#pragma unroll
        for (int kk = 0; kk < 4; ++kk) {
          ax[kk] = afrag_from_lds(h0s[p], lane, kk);
          ah[kk] = afrag_from_lds(h1s[p], lane, kk);
        }
        v8f c0 = {}, c1 = {};
#pragma unroll
        for (int kk = 0; kk < 4; ++kk) {
          c0 = __builtin_amdgcn_wmma_f32_16x16x32_f16(false, ax[kk], false, fih[0][kk], (short)0, c0, false, false);
          c1 = __builtin_amdgcn_wmma_f32_16x16x32_f16(false, ax[kk], false, fih[1][kk], (short)0, c1, false, false);
        }
#pragma unroll
        for (int kk = 0; kk < 4; ++kk) {
          c0 = __builtin_amdgcn_wmma_f32_16x16x32_f16(false, ah[kk], false, fhh[0][kk], (short)0, c0, false, false);
          c1 = __builtin_amdgcn_wmma_f32_16x16x32_f16(false, ah[kk], false, fhh[1][kk], (short)0, c1, false, false);
        }
        const int t = s - 1;
#pragma unroll
        for (int r = 0; r < 8; ++r) {
          const int m = rbase + r;
          float v0 = fast_tanh(c0[r] + bias[0]);
          float v1 = fast_tanh(c1[r] + bias[1]);
          h1s[q][m * PITCH + n_[0]] = (_Float16)v0;
          h1s[q][m * PITCH + n_[1]] = (_Float16)v1;
          out[((b0 + m) * Tdim + t) * Hdim + n_[0]] = v0;
          out[((b0 + m) * Tdim + t) * Hdim + n_[1]] = v1;
          if (s == Tdim) {                          // h_n layer 1
            hid[Bdim * Hdim + (b0 + m) * Hdim + n_[0]] = v0;
            hid[Bdim * Hdim + (b0 + m) * Hdim + n_[1]] = v1;
          }
        }
      }
    }

    // stage x_{s+1} into the other x buffer (all threads)
    {
      _Float16* d = &xs[q][xm * PITCH + xc];
#pragma unroll
      for (int j = 0; j < 4; ++j) { d[j] = (_Float16)xq0[j]; d[4 + j] = (_Float16)xq1[j]; }
    }
    __syncthreads();   // publish h0[s], h1[s-1], x[s+1]
  }
}

extern "C" void kernel_launch(void* const* d_in, const int* in_sizes, int n_in,
                              void* d_out, int out_size, void* d_ws, size_t ws_size,
                              hipStream_t stream) {
  const float* x    = (const float*)d_in[0];
  const float* Wih0 = (const float*)d_in[1];
  const float* Whh0 = (const float*)d_in[2];
  const float* bih0 = (const float*)d_in[3];
  const float* bhh0 = (const float*)d_in[4];
  const float* Wih1 = (const float*)d_in[5];
  const float* Whh1 = (const float*)d_in[6];
  const float* bih1 = (const float*)d_in[7];
  const float* bhh1 = (const float*)d_in[8];
  float* out = (float*)d_out;

  // 4 independent workgroups (16 batch rows each), 8 waves of 32 per WG.
  rnn2_pipelined_scan<<<dim3(Bdim / ROWS), dim3(256), 0, stream>>>(
      x, Wih0, Whh0, bih0, bhh0, Wih1, Whh1, bih1, bhh1, out);
}
